// CrossAttention_231928234714
// MI455X (gfx1250) — matrix operand order repaired and checked by reading, not compiled, not run on hardware
//
#include <hip/hip_runtime.h>

// ---------------------------------------------------------------------------
// CDNA5 (gfx1250) cross-attention: bf16 WMMA everywhere, f32 accumulate.
// GEMM operand staging via Tensor Data Mover (tensor_load_to_lds) with
// double-buffered LDS tiles; flash attention with online softmax.
// ---------------------------------------------------------------------------

typedef __attribute__((ext_vector_type(16))) __bf16          v16bf;
typedef __attribute__((ext_vector_type(8)))  float           v8f;
typedef __attribute__((ext_vector_type(8)))  unsigned short  u16x8;
typedef __attribute__((ext_vector_type(4)))  unsigned short  u16x4;
typedef __attribute__((ext_vector_type(4)))  unsigned int    u32x4;
typedef __attribute__((ext_vector_type(8)))  int             i32x8;
typedef __attribute__((ext_vector_type(4)))  int             i32x4;

#define NB      4
#define LQ      2048
#define LKV     2048
#define DMODEL  1024
#define DKV     768
#define NH      16
#define DH      64

#define LDS_PITCH 40   // 32 bf16 payload + 8 bf16 pad per row (80B: keeps b128 align)

__device__ __forceinline__ unsigned short f2bf(float f) {
    unsigned int u = __float_as_uint(f);
    u += 0x7fffu + ((u >> 16) & 1u);          // round-to-nearest-even
    return (unsigned short)(u >> 16);
}

union Frag { v16bf v; u16x8 h[2]; };

// A-fragment (16x32, M across lanes 0-15, K interleaved per ISA 7.12.2):
// lane (r, hl): VGPR0-3 = K[8*hl .. 8*hl+7], VGPR4-7 = K[16+8*hl .. +7]
__device__ __forceinline__ v16bf load_afrag(const unsigned short* base, int rowstride, int lane) {
    const int r = lane & 15, hl = lane >> 4;
    const unsigned short* p = base + (size_t)r * rowstride + 8 * hl;
    Frag f;
    f.h[0] = *(const u16x8*)(p);
    f.h[1] = *(const u16x8*)(p + 16);
    return f.v;
}

// B-fragment (32x16, N across lanes 0-15, K interleaved like the A-fragment):
// lane (n, hl): VGPR0-3 = Bcol[8*hl .. 8*hl+7], VGPR4-7 = Bcol[16+8*hl .. +7]; memory holds B^T rows (NT gemm).
__device__ __forceinline__ v16bf load_bfrag(const unsigned short* base, int rowstride, int lane) {
    const int n = lane & 15, hl = lane >> 4;
    const unsigned short* p = base + (size_t)n * rowstride + 8 * hl;
    Frag f;
    f.h[0] = *(const u16x8*)(p);
    f.h[1] = *(const u16x8*)(p + 16);
    return f.v;
}

#define WMMA_BF16(A, B, C) \
    __builtin_amdgcn_wmma_f32_16x16x32_bf16(false, (A), false, (B), (short)0, (C), false, false)

#define HAS_TDM __has_builtin(__builtin_amdgcn_tensor_load_to_lds)

#if HAS_TDM
// ---------------------------------------------------------------------------
// Issue one TDM 2D tile load: global (row-major, row_stride elems, bf16) ->
// LDS with 8-element pad after each 32-element row (pitch LDS_PITCH halves).
// D# layout per CDNA5 ISA ch.8 (group0 128b, group1 256b; groups 2/3 unused).
// clang-23 builtin arity: (g0 u32x4, g1 i32x8, g2 i32x4, g3 i32x4, i32x8, cpol)
// ---------------------------------------------------------------------------
__device__ __forceinline__ void tdm_load_tile_2d(unsigned lds_byte_off,
                                                 const unsigned short* gsrc,
                                                 int tile_w, int tile_h,
                                                 int row_stride) {
    unsigned long long ga = (unsigned long long)(uintptr_t)gsrc;
    u32x4 g0;
    g0[0] = 1u;                                              // count=1 (valid), user mode
    g0[1] = lds_byte_off;                                    // lds_addr [63:32]
    g0[2] = (unsigned)(ga & 0xffffffffu);                    // global_addr[31:0]
    g0[3] = (unsigned)((ga >> 32) & 0x1ffffffu) | (2u << 30);// global_addr[56:32] | type=2

    i32x8 g1;
    g1[0] = (1 << 16)        // data_size = 1 -> 2 bytes/elem
          | (1 << 20)        // pad_enable
          | (3 << 22)        // pad_interval code 3 = 16 DWORDs (one 32-bf16 row)
          | (3 << 25);       // pad_amount   code 3 = 4 DWORDs  (8 bf16 halves)
    g1[1] = (tile_w & 0xffff) << 16;                         // tensor_dim0[15:0]
    g1[2] = ((unsigned)tile_w >> 16) | ((tile_h & 0xffff) << 16); // td0[31:16] | td1[15:0]
    g1[3] = ((unsigned)tile_h >> 16) | ((tile_w & 0xffff) << 16); // td1[31:16] | tile_dim0
    g1[4] = (tile_h & 0xffff);                               // tile_dim1 | tile_dim2=0
    g1[5] = row_stride;                                      // tensor_dim0_stride[31:0]
    g1[6] = 0;                                               // stride0[47:32] | stride1[15:0]
    g1[7] = 0;                                               // stride1[47:16]

    i32x4 z4 = {0, 0, 0, 0};                                 // groups 2/3 disabled (2D)
    i32x8 z8 = {0, 0, 0, 0, 0, 0, 0, 0};
    __builtin_amdgcn_tensor_load_to_lds(g0, g1, z4, z4, z8, 0);
}
#endif

// ---------------------------------------------------------------------------
// fp32 -> bf16 cast (4 elements / thread)
// ---------------------------------------------------------------------------
__global__ __launch_bounds__(256) void cast_f32_bf16(const float* __restrict__ in,
                                                     unsigned short* __restrict__ out,
                                                     int n4) {
    int i = blockIdx.x * 256 + threadIdx.x;
    if (i < n4) {
        float4 f = ((const float4*)in)[i];
        u16x4 r = { f2bf(f.x), f2bf(f.y), f2bf(f.z), f2bf(f.w) };
        ((u16x4*)out)[i] = r;
    }
}

// ---------------------------------------------------------------------------
// NT GEMM: C[M,N] = A[M,K] * B[N,K]^T  (A,B bf16 row-major).
// 128x128 C tile / block, 8 waves (4 M-slices x 2 N-slices), each wave 32x64.
// Operand tiles DMA'd into double-buffered LDS by the Tensor Data Mover;
// wave 0 produces, TENSORcnt + workgroup barrier synchronize, compute overlaps
// the next tile's DMA.
// ---------------------------------------------------------------------------
template<int BF16_OUT, int HAS_BIAS>
__global__ __launch_bounds__(256) void gemm_nt_bf16(const unsigned short* __restrict__ A,
                                                    const unsigned short* __restrict__ Bm,
                                                    int M, int N, int K,
                                                    unsigned short* __restrict__ Cbf,
                                                    float* __restrict__ Cf,
                                                    const float* __restrict__ bias) {
    __shared__ __align__(16) unsigned short As[2][128 * LDS_PITCH];
    __shared__ __align__(16) unsigned short Bs[2][128 * LDS_PITCH];

    const int tid  = threadIdx.x;
    const int lane = tid & 31;
    const int wave = tid >> 5;
    const int wr   = wave & 3;           // 32-row slice
    const int wc   = wave >> 2;          // 64-col slice
    const int n0   = blockIdx.x * 128;
    const int m0   = blockIdx.y * 128;

    const v8f zero = {0.f,0.f,0.f,0.f,0.f,0.f,0.f,0.f};
    v8f acc[2][4] = {{zero, zero, zero, zero}, {zero, zero, zero, zero}};

    const int nk = K >> 5;               // K-steps of 32

#if HAS_TDM
    if (wave == 0) {
        tdm_load_tile_2d((unsigned)(uintptr_t)&As[0][0], A  + (size_t)m0 * K, 32, 128, K);
        tdm_load_tile_2d((unsigned)(uintptr_t)&Bs[0][0], Bm + (size_t)n0 * K, 32, 128, K);
    }
#endif

    for (int i = 0; i < nk; ++i) {
        const int buf = i & 1;
#if HAS_TDM
        if (wave == 0) {
            if (i + 1 < nk) {
                const int kt = (i + 1) * 32;
                tdm_load_tile_2d((unsigned)(uintptr_t)&As[buf ^ 1][0],
                                 A  + (size_t)m0 * K + kt, 32, 128, K);
                tdm_load_tile_2d((unsigned)(uintptr_t)&Bs[buf ^ 1][0],
                                 Bm + (size_t)n0 * K + kt, 32, 128, K);
                __builtin_amdgcn_s_wait_tensorcnt(2);   // current tile landed
            } else {
                __builtin_amdgcn_s_wait_tensorcnt(0);
            }
        }
        __syncthreads();
#else
        {   // manual staging fallback (single logical buffer)
            const int kt = i * 32;
            const int lr = tid >> 1;              // 0..127
            const int lc = (tid & 1) * 16;        // 0 / 16
            const unsigned short* ga = A  + (size_t)(m0 + lr) * K + kt + lc;
            const unsigned short* gb = Bm + (size_t)(n0 + lr) * K + kt + lc;
            *(u16x8*)&As[buf][lr * LDS_PITCH + lc]     = *(const u16x8*)ga;
            *(u16x8*)&As[buf][lr * LDS_PITCH + lc + 8] = *(const u16x8*)(ga + 8);
            *(u16x8*)&Bs[buf][lr * LDS_PITCH + lc]     = *(const u16x8*)gb;
            *(u16x8*)&Bs[buf][lr * LDS_PITCH + lc + 8] = *(const u16x8*)(gb + 8);
            __syncthreads();
        }
#endif
        v16bf a0 = load_afrag(&As[buf][(wr * 32 + 0)  * LDS_PITCH], LDS_PITCH, lane);
        v16bf a1 = load_afrag(&As[buf][(wr * 32 + 16) * LDS_PITCH], LDS_PITCH, lane);
#pragma unroll
        for (int t = 0; t < 4; ++t) {
            v16bf bf = load_bfrag(&Bs[buf][(wc * 64 + t * 16) * LDS_PITCH], LDS_PITCH, lane);
            acc[0][t] = WMMA_BF16(a0, bf, acc[0][t]);
            acc[1][t] = WMMA_BF16(a1, bf, acc[1][t]);
        }
        __syncthreads();
    }

    // C/D layout: lane(nl,hl) holds col 16t+nl, rows r*16 + v + 8*hl
    const int hl = lane >> 4;
    const int nl = lane & 15;
#pragma unroll
    for (int r = 0; r < 2; ++r) {
#pragma unroll
        for (int t = 0; t < 4; ++t) {
            const int gn = n0 + wc * 64 + t * 16 + nl;
            float badd = HAS_BIAS ? bias[gn] : 0.0f;
#pragma unroll
            for (int v = 0; v < 8; ++v) {
                const int gm = m0 + wr * 32 + r * 16 + v + 8 * hl;
                float val = acc[r][t][v] + badd;
                if (BF16_OUT) Cbf[(size_t)gm * N + gn] = f2bf(val);
                else          Cf [(size_t)gm * N + gn] = val;
            }
        }
    }
}

// ---------------------------------------------------------------------------
// Flash attention: block = 4 waves, one (b,h), 64 query rows (16/wave).
// KV streamed in 32-wide tiles, online softmax, all matmuls via bf16 WMMA.
// ---------------------------------------------------------------------------
__global__ __launch_bounds__(128) void attn_kernel(const unsigned short* __restrict__ qp,
                                                   const unsigned short* __restrict__ kp,
                                                   const unsigned short* __restrict__ vp,
                                                   unsigned short* __restrict__ outp) {
    __shared__ __align__(16) unsigned short Vt[64 * LDS_PITCH];    // V transposed: Vt[c][kvl]
    __shared__ __align__(16) unsigned short Pl[4][16 * LDS_PITCH]; // per-wave P relayout

    const int tid  = threadIdx.x;
    const int lane = tid & 31;
    const int wave = tid >> 5;
    const int nl   = lane & 15;
    const int hl   = lane >> 4;
    const int qt   = blockIdx.x;          // 0..31
    const int bh   = blockIdx.y;          // 0..63
    const int b    = bh >> 4;
    const int h    = bh & 15;
    const int q0   = qt * 64 + wave * 16;

    // Preload Q fragments for both head-dim K-steps (Dh=64 -> 2 x 32)
    const unsigned short* qbase = qp + (size_t)(b * LQ + q0) * DMODEL + h * DH;
    const v16bf qf0 = load_afrag(qbase,      DMODEL, lane);
    const v16bf qf1 = load_afrag(qbase + 32, DMODEL, lane);

    const v8f zero = {0.f,0.f,0.f,0.f,0.f,0.f,0.f,0.f};
    v8f o0 = zero, o1 = zero, o2 = zero, o3 = zero;
    float mi[8], li[8];
#pragma unroll
    for (int v = 0; v < 8; ++v) { mi[v] = -3.0e38f; li[v] = 0.0f; }

    const int vrow = tid >> 2;            // kv-local row 0..31
    const int vcol = (tid & 3) * 16;      // head-dim col chunk
    unsigned short* myP = &Pl[wave][0];

    for (int kv0 = 0; kv0 < LKV; kv0 += 32) {
        __syncthreads();
        // Stage V tile transposed into LDS (cooperative, all 4 waves)
        const unsigned short* gv = vp + (size_t)(b * LKV + kv0 + vrow) * DMODEL + h * DH + vcol;
        u16x8 va = *(const u16x8*)gv;
        u16x8 vb = *(const u16x8*)(gv + 8);
#pragma unroll
        for (int j = 0; j < 8; ++j) {
            Vt[(vcol + j)     * LDS_PITCH + vrow] = va[j];
            Vt[(vcol + 8 + j) * LDS_PITCH + vrow] = vb[j];
        }
        __syncthreads();

        // S = Q K^T for 16x32 tile (K rows are kv positions; head-dim contiguous)
        const unsigned short* kbase = kp + (size_t)(b * LKV + kv0) * DMODEL + h * DH;
        v16bf kb00 = load_bfrag(kbase,                    DMODEL, lane);
        v16bf kb01 = load_bfrag(kbase + 32,               DMODEL, lane);
        v16bf kb10 = load_bfrag(kbase + 16 * DMODEL,      DMODEL, lane);
        v16bf kb11 = load_bfrag(kbase + 16 * DMODEL + 32, DMODEL, lane);
        v8f s0 = WMMA_BF16(qf0, kb00, zero);
        s0     = WMMA_BF16(qf1, kb01, s0);
        v8f s1 = WMMA_BF16(qf0, kb10, zero);
        s1     = WMMA_BF16(qf1, kb11, s1);

        // Online softmax (rows live as v + 8*hl; reduce across 16-lane halves)
        const float scale = 0.125f;       // 1/sqrt(64)
        float a0[8], a1[8], tm[8];
#pragma unroll
        for (int v = 0; v < 8; ++v) {
            a0[v] = s0[v] * scale;
            a1[v] = s1[v] * scale;
            tm[v] = fmaxf(a0[v], a1[v]);
        }
#pragma unroll
        for (int m = 1; m <= 8; m <<= 1)
#pragma unroll
            for (int v = 0; v < 8; ++v) tm[v] = fmaxf(tm[v], __shfl_xor(tm[v], m, 32));

        float corr[8], rs[8];
#pragma unroll
        for (int v = 0; v < 8; ++v) {
            float mn = fmaxf(mi[v], tm[v]);
            corr[v] = __expf(mi[v] - mn);
            mi[v]   = mn;
            a0[v]   = __expf(a0[v] - mn);
            a1[v]   = __expf(a1[v] - mn);
            rs[v]   = a0[v] + a1[v];
        }
#pragma unroll
        for (int m = 1; m <= 8; m <<= 1)
#pragma unroll
            for (int v = 0; v < 8; ++v) rs[v] += __shfl_xor(rs[v], m, 32);
#pragma unroll
        for (int v = 0; v < 8; ++v) {
            li[v] = li[v] * corr[v] + rs[v];
            o0[v] *= corr[v]; o1[v] *= corr[v]; o2[v] *= corr[v]; o3[v] *= corr[v];
        }

        // Relayout P (C-layout -> A-layout) via per-wave LDS tile.
        // Same-wave DS ops are in-order on CDNA5; compiler barrier keeps order.
#pragma unroll
        for (int v = 0; v < 8; ++v) {
            const int m = v + 8 * hl;
            myP[m * LDS_PITCH + nl]      = f2bf(a0[v]);
            myP[m * LDS_PITCH + 16 + nl] = f2bf(a1[v]);
        }
        asm volatile("" ::: "memory");
        v16bf pf = load_afrag(myP, LDS_PITCH, lane);

        // O += P V  (B fragments from transposed V tile: contiguous in LDS)
        o0 = WMMA_BF16(pf, load_bfrag(&Vt[ 0 * LDS_PITCH], LDS_PITCH, lane), o0);
        o1 = WMMA_BF16(pf, load_bfrag(&Vt[16 * LDS_PITCH], LDS_PITCH, lane), o1);
        o2 = WMMA_BF16(pf, load_bfrag(&Vt[32 * LDS_PITCH], LDS_PITCH, lane), o2);
        o3 = WMMA_BF16(pf, load_bfrag(&Vt[48 * LDS_PITCH], LDS_PITCH, lane), o3);
    }

    // Normalize and emit bf16 merged-heads output [B*LQ, DMODEL]
#pragma unroll
    for (int v = 0; v < 8; ++v) {
        const float linv = 1.0f / li[v];
        const size_t row = (size_t)(b * LQ + q0 + v + 8 * hl) * DMODEL + h * DH + nl;
        outp[row + 0]  = f2bf(o0[v] * linv);
        outp[row + 16] = f2bf(o1[v] * linv);
        outp[row + 32] = f2bf(o2[v] * linv);
        outp[row + 48] = f2bf(o3[v] * linv);
    }
}

// ---------------------------------------------------------------------------
// Host-side orchestration
// ---------------------------------------------------------------------------
extern "C" void kernel_launch(void* const* d_in, const int* in_sizes, int n_in,
                              void* d_out, int out_size, void* d_ws, size_t ws_size,
                              hipStream_t stream) {
    (void)in_sizes; (void)n_in; (void)out_size; (void)ws_size;

    const float* q  = (const float*)d_in[0];
    const float* kv = (const float*)d_in[1];
    const float* wq = (const float*)d_in[2];
    const float* wk = (const float*)d_in[3];
    const float* wv = (const float*)d_in[4];
    const float* wo = (const float*)d_in[5];
    const float* bo = (const float*)d_in[6];
    float* out = (float*)d_out;

    char* ws = (char*)d_ws;
    size_t off = 0;
    auto alloc = [&](size_t elems) -> unsigned short* {
        unsigned short* p = (unsigned short*)(ws + off);
        off = (off + elems * 2 + 255) & ~(size_t)255;
        return p;
    };
    unsigned short* qbf  = alloc((size_t)NB * LQ * DMODEL);
    unsigned short* kvbf = alloc((size_t)NB * LKV * DKV);
    unsigned short* wqbf = alloc((size_t)DMODEL * DMODEL);
    unsigned short* wkbf = alloc((size_t)DMODEL * DKV);
    unsigned short* wvbf = alloc((size_t)DMODEL * DKV);
    unsigned short* wobf = alloc((size_t)DMODEL * DMODEL);
    unsigned short* qpbf = alloc((size_t)NB * LQ * DMODEL);
    unsigned short* kpbf = alloc((size_t)NB * LKV * DMODEL);
    unsigned short* vpbf = alloc((size_t)NB * LKV * DMODEL);
    unsigned short* abf  = alloc((size_t)NB * LQ * DMODEL);

    auto cast = [&](const float* src, unsigned short* dst, size_t n) {
        int n4 = (int)(n / 4);
        cast_f32_bf16<<<dim3((n4 + 255) / 256), dim3(256), 0, stream>>>(src, dst, n4);
    };
    cast(q,  qbf,  (size_t)NB * LQ * DMODEL);
    cast(kv, kvbf, (size_t)NB * LKV * DKV);
    cast(wq, wqbf, (size_t)DMODEL * DMODEL);
    cast(wk, wkbf, (size_t)DMODEL * DKV);
    cast(wv, wvbf, (size_t)DMODEL * DKV);
    cast(wo, wobf, (size_t)DMODEL * DMODEL);

    const dim3 blk(256);
    // Projections: qp = q wq^T, kp = kv wk^T, vp = kv wv^T  (bf16 out)
    gemm_nt_bf16<1, 0><<<dim3(DMODEL / 128, NB * LQ / 128), blk, 0, stream>>>(
        qbf, wqbf, NB * LQ, DMODEL, DMODEL, qpbf, nullptr, nullptr);
    gemm_nt_bf16<1, 0><<<dim3(DMODEL / 128, NB * LKV / 128), blk, 0, stream>>>(
        kvbf, wkbf, NB * LKV, DMODEL, DKV, kpbf, nullptr, nullptr);
    gemm_nt_bf16<1, 0><<<dim3(DMODEL / 128, NB * LKV / 128), blk, 0, stream>>>(
        kvbf, wvbf, NB * LKV, DMODEL, DKV, vpbf, nullptr, nullptr);

    // Attention (merged-head bf16 output)
    attn_kernel<<<dim3(LQ / 64, NB * NH), dim3(128), 0, stream>>>(qpbf, kpbf, vpbf, abf);

    // Output projection: out = attn wo^T + bo  (f32 out)
    gemm_nt_bf16<0, 1><<<dim3(DMODEL / 128, NB * LQ / 128), blk, 0, stream>>>(
        abf, wobf, NB * LQ, DMODEL, DMODEL, nullptr, out, bo);
}
